// LinkPredHead_46574625358326
// MI455X (gfx1250) — matrix operand order repaired and checked by reading, not compiled, not run on hardware
//
#include <hip/hip_runtime.h>
#include <hip/hip_bf16.h>
#include <math.h>

#define H 128
#define K3 384                     // 3*H
#define NEDGES 400000
#define NWAVES 8
#define TPW 8
#define TILES_PER_BLOCK (NWAVES * TPW)   // 64
#define W1_STRIDE 392              // bf16 per row of sW1 (384 + 8 pad) -> conflict-free b128
#define W2_STRIDE 136              // bf16 per row of sW2 / h1 (128 + 8 pad)
#define H2_STRIDE 36               // f32 per row of h2 (32 + 4 pad)

typedef __attribute__((ext_vector_type(16))) __bf16 v16bf;
typedef __attribute__((ext_vector_type(8)))  __bf16 v8bf;
typedef __attribute__((ext_vector_type(8)))  float  v8f;
typedef __attribute__((ext_vector_type(4)))  float  v4f;

#define SW1_BYTES (128 * W1_STRIDE * 2)          // 100352
#define SW2_BYTES (32 * W2_STRIDE * 2)           // 8704
#define SW3_BYTES (32 * 4)                       // 128
#define H1_BYTES_PER_WAVE (16 * W2_STRIDE * 2)   // 4352 (reused as f32 h2 16x36)
#define SMEM_BYTES (SW1_BYTES + SW2_BYTES + SW3_BYTES + NWAVES * H1_BYTES_PER_WAVE) // 144000

#define RELU(v) __builtin_amdgcn_fmed3f((v), 0.0f, __builtin_inff())
// pure compiler reorder fence; per-wave LDS ops are in-order in HW (ISA ch.11),
// the compiler still inserts the minimal counted data waits.
#define LDS_FENCE() asm volatile("" ::: "memory")

#define SHUF16 0,1,2,3,4,5,6,7,8,9,10,11,12,13,14,15

__global__ __launch_bounds__(NWAVES * 32, 1)
void linkpred_kernel(const float* __restrict__ x,
                     const int*   __restrict__ pei,   // [2, E]
                     const float* __restrict__ pea,   // [E, H]
                     const int*   __restrict__ nei,   // [2, E]
                     const float* __restrict__ nea,   // [E, H]
                     const float* __restrict__ W1,    // [384, 128]
                     const float* __restrict__ b1,    // [128]
                     const float* __restrict__ W2,    // [128, 25]
                     const float* __restrict__ b2,    // [25]
                     const float* __restrict__ W3,    // [25]
                     const float* __restrict__ b3,    // [1]
                     float*       __restrict__ out,   // [2*E] (pos then neg)
                     int nTiles, int blocksPerBranch)
{
    __shared__ __align__(16) unsigned char smem[SMEM_BYTES];
    __bf16* sW1     = (__bf16*)smem;
    __bf16* sW2     = (__bf16*)(smem + SW1_BYTES);
    float*  sW3     = (float*)(smem + SW1_BYTES + SW2_BYTES);
    __bf16* sH1base = (__bf16*)(smem + SW1_BYTES + SW2_BYTES + SW3_BYTES);

    const int tid  = threadIdx.x;
    const int wave = tid >> 5;
    const int lane = tid & 31;
    const int lh   = lane & 15;
    const int hi   = lane >> 4;    // 0 or 1

    // branch select (block-uniform)
    const int branch = (blockIdx.x >= blocksPerBranch) ? 1 : 0;
    const int blk    = blockIdx.x - branch * blocksPerBranch;
    const int*   eidx  = branch ? nei : pei;
    const float* eattr = branch ? nea : pea;
    float*       outp  = out + (size_t)branch * NEDGES;

    // ---- stage W1^T into LDS as bf16: sW1[n][k] = W1[k*H + n], coalesced global reads
    for (int idx = tid; idx < K3 * H; idx += NWAVES * 32) {
        int k = idx >> 7;          // 0..383
        int n = idx & (H - 1);     // 0..127
        sW1[n * W1_STRIDE + k] = (__bf16)W1[idx];
    }
    // ---- stage W2^T zero-padded to 32 columns: sW2[n][k] = (n<25) ? W2[k*25+n] : 0
    for (int idx = tid; idx < 32 * H; idx += NWAVES * 32) {
        int n = idx >> 7;          // 0..31
        int k = idx & (H - 1);     // 0..127
        float v = (n < 25) ? W2[k * 25 + n] : 0.0f;
        sW2[n * W2_STRIDE + k] = (__bf16)v;
    }
    if (tid < 32) sW3[tid] = (tid < 25) ? W3[tid] : 0.0f;
    __syncthreads();

    // per-lane bias registers
    float b1v[8];
#pragma unroll
    for (int nt = 0; nt < 8; ++nt) b1v[nt] = b1[nt * 16 + lh];
    float b2v0 = b2[lh];
    float b2v1 = ((16 + lh) < 25) ? b2[16 + lh] : 0.0f;
    float b3v  = b3[0];

    __bf16* h1 = sH1base + wave * (16 * W2_STRIDE);
    float*  h2 = (float*)h1;   // overlays h1 after layer-2 A-fragments are consumed

    const v8f zero = {0.f, 0.f, 0.f, 0.f, 0.f, 0.f, 0.f, 0.f};

    for (int t = 0; t < TPW; ++t) {
        int tile = blk * TILES_PER_BLOCK + t * NWAVES + wave;   // wave-uniform
        if (tile >= nTiles) break;

        int e    = tile * 16 + lh;        // row owned by this lane (both halves)
        int srcN = eidx[e];
        int dstN = eidx[NEDGES + e];
        const float* rp0 = x + (size_t)srcN * H;
        const float* rp1 = x + (size_t)dstN * H;
        const float* rp2 = eattr + (size_t)e * H;

        // ================= Layer 1: [16,384] @ W1 -> [16,128] =================
        v8f acc[8];
#pragma unroll
        for (int i = 0; i < 8; ++i) acc[i] = zero;

#pragma unroll
        for (int kt = 0; kt < 12; ++kt) {
            const int seg = kt >> 2;   // 0: x[src], 1: x[dst], 2: edge_attr
            const float* rp = (seg == 0) ? rp0 : ((seg == 1) ? rp1 : rp2);
            const float* p  = rp + (kt & 3) * 32 + hi * 8;
            v4f f0, f1, f2, f3;
            if (seg == 2) {            // streaming edge attrs: keep x resident in L2
                f0 = __builtin_nontemporal_load((const v4f*)p);
                f1 = __builtin_nontemporal_load((const v4f*)(p + 4));
                f2 = __builtin_nontemporal_load((const v4f*)(p + 16));
                f3 = __builtin_nontemporal_load((const v4f*)(p + 20));
            } else {
                f0 = *(const v4f*)p;        f1 = *(const v4f*)(p + 4);
                f2 = *(const v4f*)(p + 16); f3 = *(const v4f*)(p + 20);
            }
            // A fragment (16-bit A 16x32 layout: lanes<16 K blocks {0..7,16..23}, lanes>=16 +8)
            v16bf a;
#pragma unroll
            for (int i = 0; i < 4; ++i) {
                float u0 = f0[i], u1 = f1[i], u2 = f2[i], u3 = f3[i];
                if (seg < 2) {   // relu applies to node features only
                    u0 = RELU(u0); u1 = RELU(u1); u2 = RELU(u2); u3 = RELU(u3);
                }
                a[i]      = (__bf16)u0;
                a[4 + i]  = (__bf16)u1;
                a[8 + i]  = (__bf16)u2;
                a[12 + i] = (__bf16)u3;
            }
            // prefetch ALL 8 B fragments into distinct registers, then run the
            // WMMA chain back-to-back (counted waits, loads overlap the XDL pipe)
            const __bf16* bbase = sW1 + (size_t)lh * W1_STRIDE + kt * 32 + hi * 8;
            v16bf bfr[8];
#pragma unroll
            for (int nt = 0; nt < 8; ++nt) {
                const __bf16* bp = bbase + (size_t)nt * (16 * W1_STRIDE);
                v8bf blo = *(const v8bf*)bp;
                v8bf bhi = *(const v8bf*)(bp + 16);
                bfr[nt] = __builtin_shufflevector(blo, bhi, SHUF16);
            }
#pragma unroll
            for (int nt = 0; nt < 8; ++nt) {
                acc[nt] = __builtin_amdgcn_wmma_f32_16x16x32_bf16(
                    false, a, false, bfr[nt], (short)0, acc[nt], false, false);
            }
        }

        // bias + relu, spill h1 tile to LDS as bf16 (C layout: vgpr r -> M=r+8*hi, N=nt*16+lh)
#pragma unroll
        for (int nt = 0; nt < 8; ++nt) {
            float bb = b1v[nt];
#pragma unroll
            for (int r = 0; r < 8; ++r) {
                float v = RELU(acc[nt][r] + bb);
                h1[(r + hi * 8) * W2_STRIDE + nt * 16 + lh] = (__bf16)v;
            }
        }
        LDS_FENCE();

        // ================= Layer 2: [16,128] @ W2 -> [16,32(25)] =================
        // A fragments from the h1 tile; B fragments reloaded from LDS each tile
        // (keeps peak VGPR use < 256, avoiding s_set_vgpr_msb traffic)
        v16bf a2f[4];
#pragma unroll
        for (int kt = 0; kt < 4; ++kt) {
            const __bf16* ap = h1 + (size_t)lh * W2_STRIDE + kt * 32 + hi * 8;
            v8bf a0 = *(const v8bf*)ap;
            v8bf a1 = *(const v8bf*)(ap + 16);
            a2f[kt] = __builtin_shufflevector(a0, a1, SHUF16);
        }
        v16bf b2f[8];
#pragma unroll
        for (int kt = 0; kt < 4; ++kt) {
#pragma unroll
            for (int nt = 0; nt < 2; ++nt) {
                const __bf16* bp = sW2 + (size_t)(nt * 16 + lh) * W2_STRIDE + kt * 32 + hi * 8;
                v8bf blo = *(const v8bf*)bp;
                v8bf bh  = *(const v8bf*)(bp + 16);
                b2f[kt * 2 + nt] = __builtin_shufflevector(blo, bh, SHUF16);
            }
        }
        v8f acc2[2];
        acc2[0] = zero; acc2[1] = zero;
#pragma unroll
        for (int kt = 0; kt < 4; ++kt) {
#pragma unroll
            for (int nt = 0; nt < 2; ++nt) {
                acc2[nt] = __builtin_amdgcn_wmma_f32_16x16x32_bf16(
                    false, a2f[kt], false, b2f[kt * 2 + nt], (short)0, acc2[nt], false, false);
            }
        }
        LDS_FENCE();   // h1 reads issued before h2 overlays it (in-order DS pipe)

        // bias + relu, spill h2 (f32) to LDS
#pragma unroll
        for (int v = 0; v < 2; ++v) {
            float bb = v ? b2v1 : b2v0;   // zero for padded columns >= 25
#pragma unroll
            for (int r = 0; r < 8; ++r) {
                float val = RELU(acc2[v][r] + bb);
                h2[(r + hi * 8) * H2_STRIDE + v * 16 + lh] = val;
            }
        }
        LDS_FENCE();

        // ================= Layer 3: [16,25] @ W3 + sigmoid =================
        if (!hi) {   // lanes 0..15, one edge each (no WMMA under partial EXEC)
            float s = 0.f;
#pragma unroll
            for (int k = 0; k < 25; ++k)
                s += h2[lh * H2_STRIDE + k] * sW3[k];
            float z  = s + b3v;
            float sg = 1.0f / (1.0f + __expf(-z));
            __builtin_nontemporal_store(sg, outp + tile * 16 + lh);
        }
        LDS_FENCE();   // before next tile reuses h1/h2
    }
}

extern "C" void kernel_launch(void* const* d_in, const int* in_sizes, int n_in,
                              void* d_out, int out_size, void* d_ws, size_t ws_size,
                              hipStream_t stream) {
    const float* x   = (const float*)d_in[0];
    const int*   pei = (const int*)d_in[1];
    const float* pea = (const float*)d_in[2];
    const int*   nei = (const int*)d_in[3];
    const float* nea = (const float*)d_in[4];
    const float* W1  = (const float*)d_in[5];
    const float* b1  = (const float*)d_in[6];
    const float* W2  = (const float*)d_in[7];
    const float* b2  = (const float*)d_in[8];
    const float* W3  = (const float*)d_in[9];
    const float* b3  = (const float*)d_in[10];
    float* out = (float*)d_out;

    const int nTiles = (NEDGES + 15) / 16;                                    // 25000
    const int blocksPerBranch = (nTiles + TILES_PER_BLOCK - 1) / TILES_PER_BLOCK; // 391
    dim3 block(NWAVES * 32);

    linkpred_kernel<<<2 * blocksPerBranch, block, 0, stream>>>(
        x, pei, pea, nei, nea, W1, b1, W2, b2, W3, b3, out,
        nTiles, blocksPerBranch);
}
